// AudioEncoder_19619410608649
// MI455X (gfx1250) — compile-verified, matrix operand order checked by reading, not run on hardware
//
#include <hip/hip_runtime.h>
#include <hip/hip_bf16.h>
#include <math.h>

// ---------------------------------------------------------------------------
// AudioEncoder for MI455X (gfx1250): conv stem + 2 Mamba + 2 attention layers.
// All GEMM-shaped math runs on v_wmma_f32_16x16x32_f16 (f16 in, f32 accum).
// B=4, T=4096 -> L=1024, D_MODEL=256, D_INNER=512, D_STATE=16, DT_RANK=16,
// NHEAD=8, HEAD_DIM=32, D_FF=1024.  M = B*L = 4096 for all trunk GEMMs.
// ---------------------------------------------------------------------------

typedef _Float16 h16v __attribute__((ext_vector_type(16)));
typedef _Float16 h8v  __attribute__((ext_vector_type(8)));
typedef float    f8v  __attribute__((ext_vector_type(8)));
typedef float    f4v  __attribute__((ext_vector_type(4)));

#define BB      4
#define TT      4096
#define LL      1024
#define DMODEL  256
#define DINNER  512
#define DSTATE  16
#define DTRANK  16
#define NHEAD   8
#define HDIM    32
#define DFF     1024

__device__ __forceinline__ float gelu_f(float x) {
  return 0.5f * x * (1.0f + erff(x * 0.70710678118654752f));
}
__device__ __forceinline__ float silu_f(float x) {
  return x / (1.0f + __expf(-x));
}
__device__ __forceinline__ float softplus_f(float x) {
  return (x > 20.0f) ? x : log1pf(__expf(x));
}

// Load 16 consecutive f32 (16B-aligned) and store them as f16 into LDS (2x b128).
__device__ __forceinline__ void stage16_f32_to_f16(const float* __restrict__ src,
                                                   _Float16* __restrict__ dst) {
  const f4v f0 = ((const f4v*)src)[0];
  const f4v f1 = ((const f4v*)src)[1];
  const f4v f2 = ((const f4v*)src)[2];
  const f4v f3 = ((const f4v*)src)[3];
  h8v h0, h1;
  #pragma unroll
  for (int j = 0; j < 4; ++j) {
    h0[j]     = (_Float16)f0[j];
    h0[4 + j] = (_Float16)f1[j];
    h1[j]     = (_Float16)f2[j];
    h1[4 + j] = (_Float16)f3[j];
  }
  ((h8v*)dst)[0] = h0;
  ((h8v*)dst)[1] = h1;
}

// ---------------------------------------------------------------------------
// Generic WMMA GEMM: C[M,N] = act(A[M,K] * W[N,K]^T + bias) (+ residual)
// 128 threads (4 waves), 64x64 tile, K-step 32, on-the-fly f32->f16 in LDS.
// Fast vector staging path when (k0+32<=K) [and bn+64<=N for the W tile];
// guarded scalar path only for the tiny irregular GEMMs (K=16 / N=48).
// act: 0=none 1=gelu 2=silu 3=softplus
// ---------------------------------------------------------------------------
#define GP 40  // LDS row pitch in halves (80B: 16B-aligned chunks, staggered banks)

__global__ __launch_bounds__(128) void gemm_f16_wmma(
    const float* __restrict__ A, int lda,
    const float* __restrict__ W, int ldw,
    const float* __restrict__ bias,
    const float* __restrict__ residual,
    float* __restrict__ C, int ldc,
    int M, int N, int K, int act)
{
  __shared__ _Float16 sA[64 * GP];
  __shared__ _Float16 sB[64 * GP];
  const int tid  = threadIdx.x;
  const int lane = tid & 31;
  const int wave = tid >> 5;
  const int bm = blockIdx.x * 64;
  const int bn = blockIdx.y * 64;
  const int wm = (wave >> 1) * 32;   // wave quadrant
  const int wn = (wave & 1) * 32;
  const bool fullN = (bn + 64 <= N);

  f8v acc[2][2];
  #pragma unroll
  for (int i = 0; i < 2; ++i)
    #pragma unroll
    for (int j = 0; j < 2; ++j) acc[i][j] = (f8v)0.0f;

  const int r15 = lane & 15;
  const int kh  = lane >> 4;
  const int row = tid >> 1;
  const int cb  = (tid & 1) * 16;
  const int nrow = bn + row;

  for (int k0 = 0; k0 < K; k0 += 32) {
    const bool fullK = (k0 + 32 <= K);         // block-uniform
    const float* asrc = A + (size_t)(bm + row) * lda + k0 + cb;
    const float* wsrc = W + (size_t)nrow * ldw + k0 + cb;
    _Float16* adst = sA + row * GP + cb;
    _Float16* wdst = sB + row * GP + cb;

    if (fullK) {
      stage16_f32_to_f16(asrc, adst);          // 4x global_load_b128 -> 2x ds_store_b128
      if (k0 + 32 < K) __builtin_prefetch(asrc + 32, 0, 1);
    } else {
      #pragma unroll
      for (int j = 0; j < 16; ++j) {
        const int kk = k0 + cb + j;
        adst[j] = (_Float16)((kk < K) ? asrc[j] : 0.0f);
      }
    }
    if (fullK && fullN) {
      stage16_f32_to_f16(wsrc, wdst);
      if (k0 + 32 < K) __builtin_prefetch(wsrc + 32, 0, 1);
    } else {
      #pragma unroll
      for (int j = 0; j < 16; ++j) {
        const int kk = k0 + cb + j;
        wdst[j] = (_Float16)((nrow < N && kk < K) ? wsrc[j] : 0.0f);
      }
    }
    __syncthreads();

    // ---- fragments + 2x2 WMMA ----
    #pragma unroll
    for (int mi = 0; mi < 2; ++mi) {
      const _Float16* ar = sA + (wm + mi * 16 + r15) * GP;
      const h8v a0 = *(const h8v*)(ar + kh * 8);
      const h8v a1 = *(const h8v*)(ar + 16 + kh * 8);
      const h16v av = __builtin_shufflevector(a0, a1,
          0,1,2,3,4,5,6,7,8,9,10,11,12,13,14,15);
      #pragma unroll
      for (int ni = 0; ni < 2; ++ni) {
        const _Float16* br = sB + (wn + ni * 16 + r15) * GP;
        const h8v b0 = *(const h8v*)(br + kh * 8);
        const h8v b1 = *(const h8v*)(br + 16 + kh * 8);
        const h16v bv = __builtin_shufflevector(b0, b1,
            0,1,2,3,4,5,6,7,8,9,10,11,12,13,14,15);
        acc[mi][ni] = __builtin_amdgcn_wmma_f32_16x16x32_f16(
            false, av, false, bv, (short)0, acc[mi][ni], false, false);
      }
    }
    __syncthreads();
  }

  // ---- epilogue: bias -> act -> +residual -> store ----
  const int n0  = lane & 15;
  const int m0  = (lane >> 4) * 8;
  #pragma unroll
  for (int mi = 0; mi < 2; ++mi) {
    #pragma unroll
    for (int ni = 0; ni < 2; ++ni) {
      const int col = bn + wn + ni * 16 + n0;
      if (col < N) {
        const float bv = bias ? bias[col] : 0.0f;
        #pragma unroll
        for (int rr = 0; rr < 8; ++rr) {
          const int row2 = bm + wm + mi * 16 + m0 + rr;
          float v = acc[mi][ni][rr] + bv;
          if      (act == 1) v = gelu_f(v);
          else if (act == 2) v = silu_f(v);
          else if (act == 3) v = softplus_f(v);
          if (residual) v += residual[(size_t)row2 * ldc + col];
          C[(size_t)row2 * ldc + col] = v;
        }
      }
    }
  }
}

// ---------------------------------------------------------------------------
// Conv stem
// ---------------------------------------------------------------------------
__global__ void conv1_gelu_kernel(const float* __restrict__ x,
                                  const float* __restrict__ w,
                                  const float* __restrict__ bias,
                                  float* __restrict__ out)
{
  const int idx = blockIdx.x * blockDim.x + threadIdx.x;   // B*256*2048
  if (idx >= BB * DMODEL * 2048) return;
  const int l  = idx % 2048;
  const int co = (idx / 2048) % DMODEL;
  const int b  = idx / (2048 * DMODEL);
  float acc = bias[co];
  const float* wr = w + (size_t)co * 128 * 3;
  const float* xb = x + (size_t)b * 128 * TT;
  const int il0 = 2 * l - 1;
  for (int ci = 0; ci < 128; ++ci) {
    const float* xc = xb + (size_t)ci * TT;
    #pragma unroll
    for (int kk = 0; kk < 3; ++kk) {
      const int il = il0 + kk;
      if (il >= 0 && il < TT) acc += wr[ci * 3 + kk] * xc[il];
    }
  }
  out[idx] = gelu_f(acc);
}

__global__ void conv2_gelu_tr_kernel(const float* __restrict__ x,
                                     const float* __restrict__ w,
                                     const float* __restrict__ bias,
                                     float* __restrict__ out)   // (B, L, 256)
{
  const int idx = blockIdx.x * blockDim.x + threadIdx.x;   // B*L*256, co fastest
  if (idx >= BB * LL * DMODEL) return;
  const int co = idx % DMODEL;
  const int l  = (idx / DMODEL) % LL;
  const int b  = idx / (DMODEL * LL);
  float acc = bias[co];
  const float* wr = w + (size_t)co * DMODEL * 3;
  const float* xb = x + (size_t)b * DMODEL * 2048;
  const int il0 = 2 * l - 1;
  for (int ci = 0; ci < DMODEL; ++ci) {
    const float* xc = xb + (size_t)ci * 2048;
    #pragma unroll
    for (int kk = 0; kk < 3; ++kk) {
      const int il = il0 + kk;
      if (il >= 0 && il < 2048) acc += wr[ci * 3 + kk] * xc[il];
    }
  }
  out[idx] = gelu_f(acc);
}

// ---------------------------------------------------------------------------
// LayerNorm over D_MODEL=256; one wave per row.
// ---------------------------------------------------------------------------
__global__ __launch_bounds__(256) void layernorm_kernel(
    const float* __restrict__ x, const float* __restrict__ s,
    const float* __restrict__ bsh, float* __restrict__ out, int rows)
{
  const int wave = threadIdx.x >> 5;
  const int lane = threadIdx.x & 31;
  const int row  = blockIdx.x * 8 + wave;
  if (row >= rows) return;
  const float* xr = x + (size_t)row * DMODEL;
  float vals[8], sum = 0.f, sq = 0.f;
  #pragma unroll
  for (int j = 0; j < 8; ++j) {
    const float v = xr[lane + 32 * j];
    vals[j] = v; sum += v; sq += v * v;
  }
  #pragma unroll
  for (int off = 16; off; off >>= 1) {
    sum += __shfl_xor(sum, off, 32);
    sq  += __shfl_xor(sq,  off, 32);
  }
  const float mean = sum * (1.0f / DMODEL);
  const float var  = sq * (1.0f / DMODEL) - mean * mean;
  const float rstd = rsqrtf(var + 1e-5f);
  float* orow = out + (size_t)row * DMODEL;
  #pragma unroll
  for (int j = 0; j < 8; ++j) {
    const int cc = lane + 32 * j;
    orow[cc] = (vals[j] - mean) * rstd * s[cc] + bsh[cc];
  }
}

// ---------------------------------------------------------------------------
// Mamba: causal depthwise conv (K=4) + SiLU  (xz[:, :512] -> u)
// ---------------------------------------------------------------------------
__global__ void mamba_dwconv_kernel(const float* __restrict__ xz,
                                    const float* __restrict__ cw,
                                    const float* __restrict__ cb,
                                    float* __restrict__ u)
{
  const int idx = blockIdx.x * blockDim.x + threadIdx.x;   // B*L*512
  if (idx >= BB * LL * DINNER) return;
  const int d = idx % DINNER;
  const int l = (idx / DINNER) % LL;
  const int b = idx / (DINNER * LL);
  float acc = cb[d];
  #pragma unroll
  for (int kk = 0; kk < 4; ++kk) {
    const int ll = l - 3 + kk;
    if (ll >= 0)
      acc += cw[d * 4 + kk] * xz[((size_t)(b * LL + ll)) * 1024 + d];
  }
  u[idx] = silu_f(acc);
}

// ---------------------------------------------------------------------------
// Mamba selective scan: one block per batch, one thread per channel d.
// y = (scan + u*D) * silu(z)
// ---------------------------------------------------------------------------
__global__ __launch_bounds__(512) void mamba_scan_kernel(
    const float* __restrict__ xdbl,   // (B*L, 48): [dt(16) | B(16) | C(16)]
    const float* __restrict__ delta,  // (B*L, 512) softplus'ed
    const float* __restrict__ u,      // (B*L, 512)
    const float* __restrict__ xz,     // (B*L, 1024) for z = cols 512..1023
    const float* __restrict__ A_log,  // (512, 16)
    const float* __restrict__ Dp,     // (512)
    float* __restrict__ y)            // (B*L, 512)
{
  __shared__ float sB[DSTATE], sC[DSTATE];
  const int d = threadIdx.x;
  const int b = blockIdx.x;
  float Aa[DSTATE], hs[DSTATE];
  #pragma unroll
  for (int n = 0; n < DSTATE; ++n) {
    Aa[n] = -__expf(A_log[d * DSTATE + n]);
    hs[n] = 0.0f;
  }
  for (int l = 0; l < LL; ++l) {
    const size_t rb = (size_t)(b * LL + l);
    if (d < DSTATE)            sB[d]       = xdbl[rb * 48 + 16 + d];
    else if (d < 2 * DSTATE)   sC[d - 16]  = xdbl[rb * 48 + 32 + (d - 16)];
    __syncthreads();
    const float dt = delta[rb * DINNER + d];
    const float uu = u[rb * DINNER + d];
    float yv = 0.0f;
    #pragma unroll
    for (int n = 0; n < DSTATE; ++n) {
      const float dA = __expf(dt * Aa[n]);
      hs[n] = dA * hs[n] + dt * sB[n] * uu;
      yv += hs[n] * sC[n];
    }
    const float z = xz[rb * 1024 + DINNER + d];
    y[rb * DINNER + d] = (yv + uu * Dp[d]) * silu_f(z);
    __syncthreads();
  }
}

// ---------------------------------------------------------------------------
// RoPE in-place on (B, L, H, 32); one thread per (b,l,h,freq) pair.
// ---------------------------------------------------------------------------
__global__ void rope_kernel(float* __restrict__ x)
{
  const int idx = blockIdx.x * blockDim.x + threadIdx.x;   // B*L*H*16
  if (idx >= BB * LL * NHEAD * 16) return;
  int t = idx;
  const int fi  = t % 16;  t /= 16;
  const int hd  = t % NHEAD; t /= NHEAD;
  const int pos = t % LL;
  const int b   = t / LL;
  const float freq = __expf(-((float)(2 * fi) / (float)HDIM) * 9.210340371976184f); // ln(1e4)
  const float ang = (float)pos * freq;
  const float c = __cosf(ang), s = __sinf(ang);
  float* p = x + ((size_t)(b * LL + pos)) * DMODEL + hd * HDIM;
  const float x1 = p[fi], x2 = p[fi + 16];
  p[fi]      = x1 * c - x2 * s;
  p[fi + 16] = x2 * c + x1 * s;
}

// Build an A/B-layout f16 fragment from 32 consecutive f32 at ptr (16B aligned):
// elements [kh*8 .. kh*8+7] and [16+kh*8 .. 16+kh*8+7], scaled.
__device__ __forceinline__ h16v frag_from_row(const float* __restrict__ p,
                                              int kh, float scale) {
  const f4v f0 = ((const f4v*)(p + kh * 8))[0];
  const f4v f1 = ((const f4v*)(p + kh * 8))[1];
  const f4v f2 = ((const f4v*)(p + 16 + kh * 8))[0];
  const f4v f3 = ((const f4v*)(p + 16 + kh * 8))[1];
  h16v r;
  #pragma unroll
  for (int j = 0; j < 4; ++j) {
    r[j]      = (_Float16)(f0[j] * scale);
    r[4 + j]  = (_Float16)(f1[j] * scale);
    r[8 + j]  = (_Float16)(f2[j] * scale);
    r[12 + j] = (_Float16)(f3[j] * scale);
  }
  return r;
}

// ---------------------------------------------------------------------------
// Flash attention: one wave per 16-query tile per (b, head).
// Dh=32 -> one WMMA per 16x16 score tile; 32-key blocks; online softmax.
// ---------------------------------------------------------------------------
__global__ __launch_bounds__(32) void attn_flash_kernel(
    const float* __restrict__ q, const float* __restrict__ k,
    const float* __restrict__ v, float* __restrict__ o)
{
  __shared__ _Float16 pbuf[16 * GP];
  const int lane = threadIdx.x;
  const int qt = blockIdx.x, head = blockIdx.y, b = blockIdx.z;
  const int r15 = lane & 15, kh = lane >> 4;
  const float scale = 0.1767766952966369f;   // 1/sqrt(32)

  // Q fragment (A layout, scale folded in)
  const float* qp = q + ((size_t)(b * LL + qt * 16 + r15)) * DMODEL + head * HDIM;
  const h16v qa = frag_from_row(qp, kh, scale);

  f8v o0 = (f8v)0.0f, o1 = (f8v)0.0f;
  float mrow[8], lrow[8];
  #pragma unroll
  for (int rr = 0; rr < 8; ++rr) { mrow[rr] = -1e30f; lrow[rr] = 0.0f; }

  for (int kb = 0; kb < LL; kb += 32) {
    // K fragments (B layout == row-major NxK fragment of K matrix)
    const float* kp0 = k + ((size_t)(b * LL + kb + r15)) * DMODEL + head * HDIM;
    const h16v kf0 = frag_from_row(kp0, kh, 1.0f);
    const h16v kf1 = frag_from_row(kp0 + (size_t)16 * DMODEL, kh, 1.0f);
    const f8v zz = (f8v)0.0f;
    f8v s0 = __builtin_amdgcn_wmma_f32_16x16x32_f16(false, qa, false, kf0,
                                                    (short)0, zz, false, false);
    f8v s1 = __builtin_amdgcn_wmma_f32_16x16x32_f16(false, qa, false, kf1,
                                                    (short)0, zz, false, false);
    // online softmax over the 32 keys (row = kh*8+rr, spread over 16 lanes)
    #pragma unroll
    for (int rr = 0; rr < 8; ++rr) {
      const float a0 = s0[rr], a1 = s1[rr];
      float t = fmaxf(a0, a1);
      t = fmaxf(t, __shfl_xor(t, 1, 32));
      t = fmaxf(t, __shfl_xor(t, 2, 32));
      t = fmaxf(t, __shfl_xor(t, 4, 32));
      t = fmaxf(t, __shfl_xor(t, 8, 32));
      const float nm   = fmaxf(mrow[rr], t);
      const float corr = __expf(mrow[rr] - nm);
      const float p0 = __expf(a0 - nm), p1 = __expf(a1 - nm);
      float rs = p0 + p1;
      rs += __shfl_xor(rs, 1, 32);
      rs += __shfl_xor(rs, 2, 32);
      rs += __shfl_xor(rs, 4, 32);
      rs += __shfl_xor(rs, 8, 32);
      lrow[rr] = lrow[rr] * corr + rs;
      mrow[rr] = nm;
      o0[rr] *= corr;
      o1[rr] *= corr;
      const int m = kh * 8 + rr;
      pbuf[m * GP + r15]      = (_Float16)p0;
      pbuf[m * GP + 16 + r15] = (_Float16)p1;
    }
    __syncthreads();
    // P fragment (A layout) from LDS
    const h8v c0 = *(const h8v*)(pbuf + r15 * GP + kh * 8);
    const h8v c1 = *(const h8v*)(pbuf + r15 * GP + 16 + kh * 8);
    const h16v pa = __builtin_shufflevector(c0, c1,
        0,1,2,3,4,5,6,7,8,9,10,11,12,13,14,15);
    // V fragments (B layout): column n = dh, K = key index within block
    const float* vbase = v + ((size_t)(b * LL + kb)) * DMODEL + head * HDIM;
    h16v vb0, vb1;
    #pragma unroll
    for (int j = 0; j < 16; ++j) {
      const int kk = (j < 8) ? (kh * 8 + j) : (16 + kh * 8 + (j - 8));
      vb0[j] = (_Float16)vbase[(size_t)kk * DMODEL + r15];
      vb1[j] = (_Float16)vbase[(size_t)kk * DMODEL + 16 + r15];
    }
    o0 = __builtin_amdgcn_wmma_f32_16x16x32_f16(false, pa, false, vb0,
                                                (short)0, o0, false, false);
    o1 = __builtin_amdgcn_wmma_f32_16x16x32_f16(false, pa, false, vb1,
                                                (short)0, o1, false, false);
    __syncthreads();
  }
  #pragma unroll
  for (int rr = 0; rr < 8; ++rr) {
    const float inv = 1.0f / lrow[rr];
    const int row = qt * 16 + kh * 8 + rr;
    float* op = o + ((size_t)(b * LL + row)) * DMODEL + head * HDIM;
    op[r15]      = o0[rr] * inv;
    op[16 + r15] = o1[rr] * inv;
  }
}

// ---------------------------------------------------------------------------
// Host orchestration
// ---------------------------------------------------------------------------
extern "C" void kernel_launch(void* const* d_in, const int* in_sizes, int n_in,
                              void* d_out, int out_size, void* d_ws, size_t ws_size,
                              hipStream_t stream) {
  (void)in_sizes; (void)n_in; (void)out_size; (void)ws_size;
  const int M = BB * LL;                  // 4096 token rows

  // input index map (setup_inputs dict order)
  auto F = [&](int i) { return (const float*)d_in[i]; };
  const float* x       = F(0);
  const float* conv1_w = F(1); const float* conv1_b = F(2);
  const float* conv2_w = F(3); const float* conv2_b = F(4);
  // mamba layer base = 5 + 11*layer:
  //  +0 norm_s +1 norm_b +2 in_w +3 conv_w +4 conv_b +5 xproj_w
  //  +6 dt_w +7 dt_b +8 A_log +9 D +10 out_w
  // attn layer base = 27 + 16*layer:
  //  +0 n1_s +1 n1_b +2 n2_s +3 n2_b +4 q_w +5 q_b +6 k_w +7 k_b
  //  +8 v_w +9 v_b +10 o_w +11 o_b +12 f1_w +13 f1_b +14 f2_w +15 f2_b

  // workspace arena (floats); ~53 MB total
  float* ws   = (float*)d_ws;
  const size_t MEG = 1u << 20;
  float* hbuf = ws;               // 1M : residual stream (B*L*256)
  float* tmp  = ws + 1 * MEG;     // 1M : layernorm output
  float* big  = ws + 2 * MEG;     // 4M : conv1out / xz / ffn hidden
  float* r3   = ws + 6 * MEG;     // 2M : u  / q
  float* r4   = ws + 8 * MEG;     // 2M : delta / k
  float* r5   = ws + 10 * MEG;    // 2M : y  / v
  float* r6   = ws + 12 * MEG;    // 1M : attention output
  float* r7   = ws + 13 * MEG;    // 0.25M : xdbl (B*L,48)

  const dim3 gB256(256);

  // --- conv stem ---
  conv1_gelu_kernel<<<dim3((BB*DMODEL*2048 + 255) / 256), gB256, 0, stream>>>(
      x, conv1_w, conv1_b, big);
  conv2_gelu_tr_kernel<<<dim3((BB*LL*DMODEL + 255) / 256), gB256, 0, stream>>>(
      big, conv2_w, conv2_b, hbuf);

  auto gemm = [&](const float* A, int lda, const float* W, int ldw,
                  const float* bias, const float* res, float* C, int ldc,
                  int N, int K, int act) {
    gemm_f16_wmma<<<dim3(M / 64, (N + 63) / 64), dim3(128), 0, stream>>>(
        A, lda, W, ldw, bias, res, C, ldc, M, N, K, act);
  };

  // --- Mamba layers ---
  for (int layer = 0; layer < 2; ++layer) {
    const int p = 5 + 11 * layer;
    layernorm_kernel<<<dim3(M / 8), gB256, 0, stream>>>(hbuf, F(p+0), F(p+1), tmp, M);
    // in_proj: (M,256)->(M,1024), no bias
    gemm(tmp, DMODEL, F(p+2), DMODEL, nullptr, nullptr, big, 1024, 1024, DMODEL, 0);
    // depthwise causal conv + silu -> u
    mamba_dwconv_kernel<<<dim3((BB*LL*DINNER + 255) / 256), gB256, 0, stream>>>(
        big, F(p+3), F(p+4), r3);
    // x_proj: (M,512)->(M,48), no bias
    gemm(r3, DINNER, F(p+5), DINNER, nullptr, nullptr, r7, 48, 48, DINNER, 0);
    // dt_proj + softplus: (M,16)->(M,512)   (A = xdbl cols 0..15, lda=48)
    gemm(r7, 48, F(p+6), DTRANK, F(p+7), nullptr, r4, DINNER, DINNER, DTRANK, 3);
    // selective scan + D-skip + gating
    mamba_scan_kernel<<<dim3(BB), dim3(DINNER), 0, stream>>>(
        r7, r4, r3, big, F(p+8), F(p+9), r5);
    // out_proj + residual: h = h + y @ out_w^T
    gemm(r5, DINNER, F(p+10), DINNER, nullptr, hbuf, hbuf, DMODEL, DMODEL, DINNER, 0);
  }

  // --- Attention layers ---
  for (int layer = 0; layer < 2; ++layer) {
    const int p = 27 + 16 * layer;
    layernorm_kernel<<<dim3(M / 8), gB256, 0, stream>>>(hbuf, F(p+0), F(p+1), tmp, M);
    gemm(tmp, DMODEL, F(p+4), DMODEL, F(p+5), nullptr, r3, DMODEL, DMODEL, DMODEL, 0); // q
    gemm(tmp, DMODEL, F(p+6), DMODEL, F(p+7), nullptr, r4, DMODEL, DMODEL, DMODEL, 0); // k
    gemm(tmp, DMODEL, F(p+8), DMODEL, F(p+9), nullptr, r5, DMODEL, DMODEL, DMODEL, 0); // v
    rope_kernel<<<dim3((BB*LL*NHEAD*16 + 255) / 256), gB256, 0, stream>>>(r3);
    rope_kernel<<<dim3((BB*LL*NHEAD*16 + 255) / 256), gB256, 0, stream>>>(r4);
    attn_flash_kernel<<<dim3(LL / 16, NHEAD, BB), dim3(32), 0, stream>>>(r3, r4, r5, r6);
    // o_proj + residual
    gemm(r6, DMODEL, F(p+10), DMODEL, F(p+11), hbuf, hbuf, DMODEL, DMODEL, DMODEL, 0);
    // FFN
    layernorm_kernel<<<dim3(M / 8), gB256, 0, stream>>>(hbuf, F(p+2), F(p+3), tmp, M);
    gemm(tmp, DMODEL, F(p+12), DMODEL, F(p+13), nullptr, big, DFF, DFF, DMODEL, 1); // gelu
    float* outp = (layer == 1) ? (float*)d_out : hbuf;
    gemm(big, DFF, F(p+14), DFF, F(p+15), hbuf, outp, DMODEL, DMODEL, DFF, 0);
  }
}
